// decoder_mix_68487548502138
// MI455X (gfx1250) — matrix operand order repaired and checked by reading, not compiled, not run on hardware
//
#include <hip/hip_runtime.h>
#include <hip/hip_bf16.h>

typedef _Float16 h8   __attribute__((ext_vector_type(8)));
typedef _Float16 v16h __attribute__((ext_vector_type(16)));
typedef float    v8f  __attribute__((ext_vector_type(8)));
typedef float    f4   __attribute__((ext_vector_type(4)));

#define N_MIX 16
#define BATCH 16384
#define Z_DIM 64
#define H_DIM 256
#define X_DIM 784

#define TILE_B 64
#define ZSTR   (Z_DIM + 8)    // 72 halves: padded LDS row stride for z
#define HSTR   (H_DIM + 8)    // 264 halves: padded LDS row stride for h1/h2
#define SMEM_BYTES ((TILE_B * ZSTR + 2 * TILE_B * HSTR) * 2)

union AFrag { v16h v; h8 h[2]; };

// -------- one-shot prep: W [mix][K][N] f32  ->  Wt [mix][N][K] f16 --------
__global__ void transpose_to_f16(const float* __restrict__ W,
                                 _Float16* __restrict__ Wt,
                                 int K, int N) {
  size_t idx   = (size_t)blockIdx.x * blockDim.x + threadIdx.x;
  size_t total = (size_t)N_MIX * K * N;
  if (idx >= total) return;
  int    k    = (int)(idx % K);
  int    nIdx = (int)((idx / K) % N);
  size_t mix  = idx / ((size_t)K * N);
  Wt[idx] = (_Float16)W[(mix * K + k) * N + nIdx];
}

// A fragment (16x32 f16): lane holds row M=lane%16; lanes<16: K=kb+0..7 & kb+16..23,
// lanes>=16: K=kb+8..15 & kb+24..31  (ISA 16-bit A layout)
__device__ __forceinline__ v16h load_a_lds(const _Float16* base, int stride,
                                           int kb, int lane) {
  int r    = lane & 15;
  int koff = kb + ((lane >> 4) << 3);
  const _Float16* p = base + r * stride + koff;
  AFrag a;
  a.h[0] = *(const h8*)(p);
  a.h[1] = *(const h8*)(p + 16);
  return a.v;
}

// B fragment (32x16 f16): lane holds column N=lane%16; lanes<16: K=kb+0..7 & kb+16..23,
// lanes>=16: K=kb+8..15 & kb+24..31 — from [N][K]-major Wt (matches A layout).
__device__ __forceinline__ v16h load_b_wt(const _Float16* WtCols, int K,
                                          int kb, int lane) {
  int c  = lane & 15;
  int k0 = kb + ((lane >> 4) << 3);
  const _Float16* p = WtCols + (size_t)c * K + k0;
  AFrag b;
  b.h[0] = *(const h8*)(p);
  b.h[1] = *(const h8*)(p + 16);
  return b.v;
}

__device__ __forceinline__ v8f wmma_f16(v16h a, v16h b, v8f c) {
  return __builtin_amdgcn_wmma_f32_16x16x32_f16(false, a, false, b, (short)0, c,
                                                false, false);
}

__global__ __launch_bounds__(256)
void decoder_mix_kernel(const float* __restrict__ s,
                        const _Float16* __restrict__ Wt1, const float* __restrict__ b1,
                        const _Float16* __restrict__ Wt2, const float* __restrict__ b2,
                        const _Float16* __restrict__ Wt3, const float* __restrict__ b3,
                        float* __restrict__ out) {
  extern __shared__ _Float16 smem[];
  _Float16* zbuf = smem;                        // [64][ZSTR] f16
  _Float16* h1   = smem + TILE_B * ZSTR;        // [64][HSTR] f16
  _Float16* h2   = h1 + TILE_B * HSTR;          // [64][HSTR] f16

  const int mix  = blockIdx.y;
  const int row0 = blockIdx.x * TILE_B;
  const int tid  = (int)threadIdx.x;
  const int wave = tid >> 5;
  const int lane = tid & 31;
  const int col  = lane & 15;         // N within tile (C/D layout)
  const int rhi  = (lane >> 4) << 3;  // +8 row offset for upper lane half

  // ---- stage z tile [64][64] into LDS, f32 -> f16 ----
  {
    const float* zsrc = s + ((size_t)mix * BATCH + row0) * Z_DIM;
    int r  = tid >> 2;
    int c0 = (tid & 3) * 16;
    const f4* src = (const f4*)(zsrc + (size_t)r * Z_DIM + c0);
    _Float16* dst = zbuf + r * ZSTR + c0;
#pragma unroll
    for (int i = 0; i < 4; ++i) {
      f4 v = src[i];
      dst[i * 4 + 0] = (_Float16)v.x;
      dst[i * 4 + 1] = (_Float16)v.y;
      dst[i * 4 + 2] = (_Float16)v.z;
      dst[i * 4 + 3] = (_Float16)v.w;
    }
  }
  __syncthreads();

  // ================= layer 1: h1 = relu(z @ W1 + b1) =================
  {
    const _Float16* w = Wt1 + (size_t)mix * H_DIM * Z_DIM;   // [256][64]
    const float*    b = b1 + (size_t)mix * H_DIM;
    const int nc0 = wave * 2;
    v8f acc[2][4] = {};
#pragma unroll
    for (int ks = 0; ks < Z_DIM / 32; ++ks) {
      v16h bf0 = load_b_wt(w + (size_t)(nc0 + 0) * 16 * Z_DIM, Z_DIM, ks * 32, lane);
      v16h bf1 = load_b_wt(w + (size_t)(nc0 + 1) * 16 * Z_DIM, Z_DIM, ks * 32, lane);
#pragma unroll
      for (int m = 0; m < 4; ++m) {
        v16h af = load_a_lds(zbuf + m * 16 * ZSTR, ZSTR, ks * 32, lane);
        acc[0][m] = wmma_f16(af, bf0, acc[0][m]);
        acc[1][m] = wmma_f16(af, bf1, acc[1][m]);
      }
    }
#pragma unroll
    for (int t = 0; t < 2; ++t) {
      const int nc = nc0 + t;
      const float bias = b[nc * 16 + col];
#pragma unroll
      for (int m = 0; m < 4; ++m)
#pragma unroll
        for (int r = 0; r < 8; ++r) {
          float v = fmaxf(acc[t][m][r] + bias, 0.0f);
          h1[(m * 16 + rhi + r) * HSTR + nc * 16 + col] = (_Float16)v;
        }
    }
  }
  __syncthreads();

  // ================= layer 2: h2 = relu(h1 @ W2 + b2) =================
  {
    const _Float16* w = Wt2 + (size_t)mix * H_DIM * H_DIM;   // [256][256]
    const float*    b = b2 + (size_t)mix * H_DIM;
    const int nc0 = wave * 2;
    v8f acc[2][4] = {};
#pragma unroll 2
    for (int ks = 0; ks < H_DIM / 32; ++ks) {
      v16h bf0 = load_b_wt(w + (size_t)(nc0 + 0) * 16 * H_DIM, H_DIM, ks * 32, lane);
      v16h bf1 = load_b_wt(w + (size_t)(nc0 + 1) * 16 * H_DIM, H_DIM, ks * 32, lane);
#pragma unroll
      for (int m = 0; m < 4; ++m) {
        v16h af = load_a_lds(h1 + m * 16 * HSTR, HSTR, ks * 32, lane);
        acc[0][m] = wmma_f16(af, bf0, acc[0][m]);
        acc[1][m] = wmma_f16(af, bf1, acc[1][m]);
      }
    }
#pragma unroll
    for (int t = 0; t < 2; ++t) {
      const int nc = nc0 + t;
      const float bias = b[nc * 16 + col];
#pragma unroll
      for (int m = 0; m < 4; ++m)
#pragma unroll
        for (int r = 0; r < 8; ++r) {
          float v = fmaxf(acc[t][m][r] + bias, 0.0f);
          h2[(m * 16 + rhi + r) * HSTR + nc * 16 + col] = (_Float16)v;
        }
    }
  }
  __syncthreads();

  // ================= layer 3: x = sigmoid(h2 @ W3 + b3) =================
  {
    const _Float16* w = Wt3 + (size_t)mix * (size_t)X_DIM * H_DIM;  // [784][256]
    const float*    b = b3 + (size_t)mix * X_DIM;
    float* om = out + ((size_t)mix * BATCH + row0) * X_DIM;

    int nc = wave;
    // pairs (nc, nc+8): wave-uniform control flow
    for (; nc + 8 < X_DIM / 16; nc += 16) {
      v8f acc[2][4] = {};
#pragma unroll 2
      for (int ks = 0; ks < H_DIM / 32; ++ks) {
        v16h bf0 = load_b_wt(w + (size_t)(nc + 0) * 16 * H_DIM, H_DIM, ks * 32, lane);
        v16h bf1 = load_b_wt(w + (size_t)(nc + 8) * 16 * H_DIM, H_DIM, ks * 32, lane);
#pragma unroll
        for (int m = 0; m < 4; ++m) {
          v16h af = load_a_lds(h2 + m * 16 * HSTR, HSTR, ks * 32, lane);
          acc[0][m] = wmma_f16(af, bf0, acc[0][m]);
          acc[1][m] = wmma_f16(af, bf1, acc[1][m]);
        }
      }
#pragma unroll
      for (int t = 0; t < 2; ++t) {
        const int nct = nc + t * 8;
        const float bias = b[nct * 16 + col];
#pragma unroll
        for (int m = 0; m < 4; ++m)
#pragma unroll
          for (int r = 0; r < 8; ++r) {
            float v = acc[t][m][r] + bias;
            float sg = __builtin_amdgcn_rcpf(1.0f + __expf(-v));
            om[(size_t)(m * 16 + rhi + r) * X_DIM + nct * 16 + col] = sg;
          }
      }
    }
    // single-tile tail (only wave 0 reaches; uniform within wave)
    for (; nc < X_DIM / 16; nc += 16) {
      v8f acc[4] = {};
#pragma unroll 2
      for (int ks = 0; ks < H_DIM / 32; ++ks) {
        v16h bf = load_b_wt(w + (size_t)nc * 16 * H_DIM, H_DIM, ks * 32, lane);
#pragma unroll
        for (int m = 0; m < 4; ++m) {
          v16h af = load_a_lds(h2 + m * 16 * HSTR, HSTR, ks * 32, lane);
          acc[m] = wmma_f16(af, bf, acc[m]);
        }
      }
      const float bias = b[nc * 16 + col];
#pragma unroll
      for (int m = 0; m < 4; ++m)
#pragma unroll
        for (int r = 0; r < 8; ++r) {
          float v = acc[m][r] + bias;
          float sg = __builtin_amdgcn_rcpf(1.0f + __expf(-v));
          om[(size_t)(m * 16 + rhi + r) * X_DIM + nc * 16 + col] = sg;
        }
    }
  }
}

extern "C" void kernel_launch(void* const* d_in, const int* in_sizes, int n_in,
                              void* d_out, int out_size, void* d_ws, size_t ws_size,
                              hipStream_t stream) {
  (void)in_sizes; (void)n_in; (void)out_size; (void)ws_size;
  const float* s  = (const float*)d_in[0];
  const float* W1 = (const float*)d_in[1];
  const float* b1 = (const float*)d_in[2];
  const float* W2 = (const float*)d_in[3];
  const float* b2 = (const float*)d_in[4];
  const float* W3 = (const float*)d_in[5];
  const float* b3 = (const float*)d_in[6];
  float* out = (float*)d_out;

  _Float16* wt1 = (_Float16*)d_ws;                              // 16*256*64  f16
  _Float16* wt2 = wt1 + (size_t)N_MIX * H_DIM * Z_DIM;          // 16*256*256 f16
  _Float16* wt3 = wt2 + (size_t)N_MIX * H_DIM * H_DIM;          // 16*784*256 f16

  // prep: transpose + f32->f16 (weights ~9 MB f16, L2-resident afterwards)
  transpose_to_f16<<<(N_MIX * Z_DIM * H_DIM + 255) / 256, 256, 0, stream>>>(W1, wt1, Z_DIM, H_DIM);
  transpose_to_f16<<<(N_MIX * H_DIM * H_DIM + 255) / 256, 256, 0, stream>>>(W2, wt2, H_DIM, H_DIM);
  transpose_to_f16<<<(N_MIX * H_DIM * X_DIM + 255) / 256, 256, 0, stream>>>(W3, wt3, H_DIM, X_DIM);

  dim3 grid(BATCH / TILE_B, N_MIX);
  decoder_mix_kernel<<<grid, 256, SMEM_BYTES, stream>>>(s, wt1, b1, wt2, b2, wt3, b3, out);
}
